// RNN_57595511439401
// MI455X (gfx1250) — matrix operand order repaired and checked by reading, not compiled, not run on hardware
//
#include <hip/hip_runtime.h>

// ---------------------------------------------------------------------------
// RNN teacher-forced likelihood on gfx1250 (MI455X).
// bf16 WMMA GEMMs (v_wmma_f32_16x16x32_bf16), double-buffered LDS tiles filled
// with GLOBAL_LOAD_ASYNC_TO_LDS (ASYNCcnt pipeline) when available.
// ---------------------------------------------------------------------------

typedef __attribute__((ext_vector_type(16))) __bf16 v16bf;
typedef __attribute__((ext_vector_type(8)))  float  v8f;
typedef __attribute__((ext_vector_type(4)))  int    v4i;

union FragBF {
    uint4 u[2];
    v16bf v;
};

#define BDIM 256

#if defined(__has_builtin)
#if __has_builtin(__builtin_amdgcn_global_load_async_to_lds_b128) && \
    __has_builtin(__builtin_amdgcn_s_wait_asynccnt)
#define USE_ASYNC_LDS 1
#endif
#endif
#ifndef USE_ASYNC_LDS
#define USE_ASYNC_LDS 0
#endif

// builtin signature (from hipcc diagnostic): arg0 = AS(1) v4i*, arg1 = AS(3) v4i*
#define AS1I4(p) ((__attribute__((address_space(1))) v4i*)(p))
#define AS3I4(p) ((__attribute__((address_space(3))) v4i*)(p))

// ---------------------------------------------------------------------------
// Generic C = act(A * W^T + bias). A:[M,K] bf16 row-major (optionally gathered
// rows), W:[N,K] bf16 row-major. Outputs fp32 and/or bf16 (ldc = N).
// Grid: (N/128, M/128), block 256 (8 waves). Requires M%128==N%128==K%32==0.
// ---------------------------------------------------------------------------
__global__ __launch_bounds__(BDIM)
void gemm_bf16_wmma_kernel(const __bf16* __restrict__ A, int lda,
                           const int* __restrict__ gather,
                           const __bf16* __restrict__ W, int ldw,
                           const float* __restrict__ bias,
                           float* __restrict__ Cf, __bf16* __restrict__ Cbf,
                           int N, int K, int relu) {
    // double-buffered tiles, padded stride 40 halves = 80B (16B aligned rows)
    __shared__ __bf16 As[2][128][40];
    __shared__ __bf16 Bs[2][128][40];

    const int tid  = threadIdx.x;
    const int row0 = blockIdx.y * 128;
    const int col0 = blockIdx.x * 128;
    const int wave = tid >> 5;
    const int lane = tid & 31;
    const int wm   = wave >> 1;       // 0..3  -> 32-row strip
    const int wn   = wave & 1;        // 0..1  -> 64-col strip
    const int l16  = lane & 15;
    const bool hiL = (lane >= 16);

    // ---- loop-invariant staging coordinates (gather hoisted out of K loop)
    const int rT = tid >> 2;              // 0..63 : tile row (this thread, chunk 0)
    const int co = (tid & 3) * 8;         // half offset within 32-half row
    int ga0 = row0 + rT, ga1 = row0 + 64 + rT;
    if (gather) { ga0 = gather[ga0]; ga1 = gather[ga1]; }
    const __bf16* apg[2] = { A + (size_t)ga0 * lda + co,
                             A + (size_t)ga1 * lda + co };
    const __bf16* wpg[2] = { W + (size_t)(col0 + rT) * ldw + co,
                             W + (size_t)(col0 + 64 + rT) * ldw + co };

    v8f acc[2][4] = {};
    const int nk = K >> 5;

    // issue one 128x32 A tile + 128x32 W tile (4 x 16B chunks per thread)
    auto issue_tile = [&](int buf, int kofs) {
#pragma unroll
        for (int i = 0; i < 2; ++i) {
            const __bf16* gpA = apg[i] + kofs;
            const __bf16* gpW = wpg[i] + kofs;
            void* ldA = (void*)&As[buf][rT + i * 64][co];
            void* ldW = (void*)&Bs[buf][rT + i * 64][co];
#if USE_ASYNC_LDS
            __builtin_amdgcn_global_load_async_to_lds_b128(AS1I4(gpA), AS3I4(ldA), 0, 0);
            __builtin_amdgcn_global_load_async_to_lds_b128(AS1I4(gpW), AS3I4(ldW), 0, 0);
#else
            *(uint4*)ldA = *(const uint4*)gpA;
            *(uint4*)ldW = *(const uint4*)gpW;
            __builtin_prefetch(gpA + 64, 0, 3);   // next+1 K tile, near scope
            __builtin_prefetch(gpW + 64, 0, 3);
#endif
        }
    };

    issue_tile(0, 0);

    for (int kt = 0; kt < nk; ++kt) {
        const int buf = kt & 1;
        if (kt + 1 < nk) {
            issue_tile(buf ^ 1, (kt + 1) * 32);   // fill other buffer
#if USE_ASYNC_LDS
            __builtin_amdgcn_s_wait_asynccnt(4);  // tile kt landed; kt+1 in flight
#endif
        } else {
#if USE_ASYNC_LDS
            __builtin_amdgcn_s_wait_asynccnt(0);
#endif
        }
        __syncthreads();

        // ---- load fragments (ISA 7.12.2 layouts) -------------------------
        FragBF a[2], b[4];
        {
            const int ko = hiL ? 8 : 0;   // A: lane<16 K0-7/K16-23 ; else K8-15/K24-31
#pragma unroll
            for (int f = 0; f < 2; ++f) {
                const int r = wm * 32 + f * 16 + l16;
                a[f].u[0] = *(const uint4*)&As[buf][r][ko];
                a[f].u[1] = *(const uint4*)&As[buf][r][ko + 16];
            }
            const int kb = hiL ? 8 : 0;   // B: lane<16 K0-7/K16-23 ; else K8-15/K24-31
#pragma unroll
            for (int g = 0; g < 4; ++g) {
                const int n = wn * 64 + g * 16 + l16;
                b[g].u[0] = *(const uint4*)&Bs[buf][n][kb];
                b[g].u[1] = *(const uint4*)&Bs[buf][n][kb + 16];
            }
        }

#pragma unroll
        for (int f = 0; f < 2; ++f)
#pragma unroll
            for (int g = 0; g < 4; ++g)
                acc[f][g] = __builtin_amdgcn_wmma_f32_16x16x32_bf16(
                    false, a[f].v, false, b[g].v, (short)0, acc[f][g],
                    false, false);
        __syncthreads();   // all waves done with buf before it is refilled
    }

    // ---- epilogue: bias, optional relu, fp32 + bf16 stores ---------------
#pragma unroll
    for (int f = 0; f < 2; ++f) {
#pragma unroll
        for (int g = 0; g < 4; ++g) {
            const int col  = col0 + wn * 64 + g * 16 + l16;
            const float bv = bias ? bias[col] : 0.0f;
#pragma unroll
            for (int r = 0; r < 8; ++r) {
                const int row = row0 + wm * 32 + f * 16 + (hiL ? 8 : 0) + r;
                float v = acc[f][g][r] + bv;
                if (relu) v = fmaxf(v, 0.0f);
                if (Cf)  Cf[(size_t)row * N + col]  = v;
                if (Cbf) Cbf[(size_t)row * N + col] = (__bf16)v;
            }
        }
    }
}

// ---------------------------------------------------------------------------
// fp32 -> bf16 cast
// ---------------------------------------------------------------------------
__global__ void cast_f32_bf16_kernel(const float* __restrict__ s,
                                     __bf16* __restrict__ d, int n) {
    int i = blockIdx.x * BDIM + threadIdx.x;
    if (i < n) d[i] = (__bf16)s[i];
}

// ---------------------------------------------------------------------------
// gather table: gidx[t*B + b] = (t==0) ? <s> : target[b, t-1]
// ---------------------------------------------------------------------------
__global__ void build_gather_kernel(const int* __restrict__ target,
                                    int* __restrict__ gidx) {
    const int g = blockIdx.x * BDIM + threadIdx.x;   // g = t*256 + b
    if (g >= 128 * 256) return;
    const int t = g >> 8;
    const int b = g & 255;
    gidx[g] = (t == 0) ? 0 : target[(size_t)b * 128 + (t - 1)];
}

// ---------------------------------------------------------------------------
// Fused GRU gates + residual + LayerNorm for one cell, one block per batch row
// ---------------------------------------------------------------------------
__global__ __launch_bounds__(BDIM)
void gru_gate_ln_kernel(const float* __restrict__ gi, const float* __restrict__ gh,
                        const float* __restrict__ hprev, const float* __restrict__ xin,
                        const float* __restrict__ lns, const float* __restrict__ lnb,
                        float* __restrict__ hf, __bf16* __restrict__ hbf) {
    constexpr int Hh = 1024;
    __shared__ float s[Hh];
    __shared__ float r1[BDIM], r2[BDIM];
    const int b   = blockIdx.x;
    const int tid = threadIdx.x;

    float sum = 0.0f, sq = 0.0f;
    for (int j = tid; j < Hh; j += BDIM) {
        const size_t g3 = (size_t)b * 3 * Hh;
        float ir = gi[g3 + j],         hr = gh[g3 + j];
        float iz = gi[g3 + Hh + j],    hz = gh[g3 + Hh + j];
        float in_ = gi[g3 + 2*Hh + j], hn = gh[g3 + 2*Hh + j];
        float r = 1.0f / (1.0f + __expf(-(ir + hr)));
        float z = 1.0f / (1.0f + __expf(-(iz + hz)));
        float n = tanhf(in_ + r * hn);
        float hnew = (1.0f - z) * n + z * hprev[(size_t)b * Hh + j];
        float v = hnew + xin[(size_t)b * Hh + j];     // residual before LN
        s[j] = v;  sum += v;  sq += v * v;
    }
    r1[tid] = sum; r2[tid] = sq;
    __syncthreads();
    for (int st = BDIM / 2; st > 0; st >>= 1) {
        if (tid < st) { r1[tid] += r1[tid + st]; r2[tid] += r2[tid + st]; }
        __syncthreads();
    }
    const float mu   = r1[0] * (1.0f / Hh);
    const float var  = r2[0] * (1.0f / Hh) - mu * mu;
    const float rstd = rsqrtf(var + 1e-5f);
    for (int j = tid; j < Hh; j += BDIM) {
        float v = (s[j] - mu) * rstd * lns[j] + lnb[j];
        hf[(size_t)b * Hh + j]  = v;
        hbf[(size_t)b * Hh + j] = (__bf16)v;
    }
}

// ---------------------------------------------------------------------------
// log-softmax over V=512 + accumulate log-prob of target token
// ---------------------------------------------------------------------------
__global__ __launch_bounds__(BDIM)
void logsm_acc_kernel(const float* __restrict__ logits,
                      const int* __restrict__ target, int t,
                      float* __restrict__ out) {
    constexpr int Vv = 512, Tt = 128;
    __shared__ float red[BDIM];
    const int b = blockIdx.x, tid = threadIdx.x;

    float m = -1e30f;
    for (int j = tid; j < Vv; j += BDIM) m = fmaxf(m, logits[(size_t)b * Vv + j]);
    red[tid] = m; __syncthreads();
    for (int st = BDIM / 2; st > 0; st >>= 1) {
        if (tid < st) red[tid] = fmaxf(red[tid], red[tid + st]);
        __syncthreads();
    }
    m = red[0];
    __syncthreads();

    float ssum = 0.0f;
    for (int j = tid; j < Vv; j += BDIM) ssum += __expf(logits[(size_t)b * Vv + j] - m);
    red[tid] = ssum; __syncthreads();
    for (int st = BDIM / 2; st > 0; st >>= 1) {
        if (tid < st) red[tid] += red[tid + st];
        __syncthreads();
    }
    if (tid == 0) {
        const int tgt = target[(size_t)b * Tt + t];
        out[b] += logits[(size_t)b * Vv + tgt] - m - __logf(red[0]);
    }
}

// ---------------------------------------------------------------------------
// Host orchestration
// ---------------------------------------------------------------------------
extern "C" void kernel_launch(void* const* d_in, const int* in_sizes, int n_in,
                              void* d_out, int out_size, void* d_ws, size_t ws_size,
                              hipStream_t stream) {
    (void)in_sizes; (void)n_in; (void)out_size; (void)ws_size;
    constexpr int B = 256, T = 128, V = 512, E = 512, H = 1024;

    const int*   target = (const int*)  d_in[0];
    const float* emb    = (const float*)d_in[1];
    const float* projW  = (const float*)d_in[2];
    const float* projb  = (const float*)d_in[3];
    const float* Wih    = (const float*)d_in[4];
    const float* Whh    = (const float*)d_in[5];
    const float* bih    = (const float*)d_in[6];
    const float* bhh    = (const float*)d_in[7];
    const float* ln_s   = (const float*)d_in[8];
    const float* ln_b   = (const float*)d_in[9];
    const float* fc1W   = (const float*)d_in[10];
    const float* fc1b   = (const float*)d_in[11];
    const float* fc2W   = (const float*)d_in[12];
    const float* fc2b   = (const float*)d_in[13];
    float*       out    = (float*)d_out;

    // ---- carve workspace (all offsets 256B aligned) ----------------------
    size_t off = 0;
    auto carve = [&](size_t bytes) -> void* {
        void* p = (char*)d_ws + off;
        off += (bytes + 255) & ~(size_t)255;
        return p;
    };
    __bf16* emb_bf   = (__bf16*)carve((size_t)V * E * 2);
    __bf16* projW_bf = (__bf16*)carve((size_t)H * E * 2);
    __bf16* Wih_bf   = (__bf16*)carve((size_t)3 * 3 * H * H * 2);
    __bf16* Whh_bf   = (__bf16*)carve((size_t)3 * 3 * H * H * 2);
    __bf16* fc1W_bf  = (__bf16*)carve((size_t)2 * H * H * 2);
    __bf16* fc2W_bf  = (__bf16*)carve((size_t)V * 2 * H * 2);
    float*  x_f      = (float*) carve((size_t)B * H * 4);
    __bf16* x_bf     = (__bf16*)carve((size_t)B * H * 2);
    float*  h_f      = (float*) carve((size_t)3 * B * H * 4);
    __bf16* h_bf     = (__bf16*)carve((size_t)3 * B * H * 2);
    float*  gi_f     = (float*) carve((size_t)B * 3 * H * 4);
    float*  gh_f     = (float*) carve((size_t)B * 3 * H * 4);
    __bf16* y1_bf    = (__bf16*)carve((size_t)B * 2 * H * 2);
    float*  logits   = (float*) carve((size_t)B * V * 4);
    int*    gidx     = (int*)   carve((size_t)T * B * 4);

    auto cast = [&](const float* s, __bf16* d, int n) {
        cast_f32_bf16_kernel<<<(n + BDIM - 1) / BDIM, BDIM, 0, stream>>>(s, d, n);
    };
    cast(emb,   emb_bf,   V * E);
    cast(projW, projW_bf, H * E);
    cast(Wih,   Wih_bf,   3 * 3 * H * H);
    cast(Whh,   Whh_bf,   3 * 3 * H * H);
    cast(fc1W,  fc1W_bf,  2 * H * H);
    cast(fc2W,  fc2W_bf,  V * 2 * H);
    build_gather_kernel<<<(T * B + BDIM - 1) / BDIM, BDIM, 0, stream>>>(target, gidx);

    (void)hipMemsetAsync(h_f,  0, (size_t)3 * B * H * 4, stream);
    (void)hipMemsetAsync(h_bf, 0, (size_t)3 * B * H * 2, stream);
    (void)hipMemsetAsync(out,  0, (size_t)B * 4, stream);

    for (int t = 0; t < T; ++t) {
        // x_t = emb[tok_t] @ projW^T + projb
        gemm_bf16_wmma_kernel<<<dim3(H / 128, B / 128), BDIM, 0, stream>>>(
            emb_bf, E, gidx + (size_t)t * B, projW_bf, E, projb,
            x_f, x_bf, H, E, 0);

        for (int i = 0; i < 3; ++i) {
            const __bf16* xin_bf = (i == 0) ? x_bf : (h_bf + (size_t)(i - 1) * B * H);
            const float*  xin_f  = (i == 0) ? x_f  : (h_f  + (size_t)(i - 1) * B * H);
            // gh = h_prev @ Whh^T + bhh   (reads h_bf[i] from previous step)
            gemm_bf16_wmma_kernel<<<dim3(3 * H / 128, B / 128), BDIM, 0, stream>>>(
                h_bf + (size_t)i * B * H, H, nullptr,
                Whh_bf + (size_t)i * 3 * H * H, H, bhh + (size_t)i * 3 * H,
                gh_f, nullptr, 3 * H, H, 0);
            // gi = x @ Wih^T + bih
            gemm_bf16_wmma_kernel<<<dim3(3 * H / 128, B / 128), BDIM, 0, stream>>>(
                xin_bf, H, nullptr,
                Wih_bf + (size_t)i * 3 * H * H, H, bih + (size_t)i * 3 * H,
                gi_f, nullptr, 3 * H, H, 0);
            // gates + residual + LN -> h[i] (fp32 + bf16)
            gru_gate_ln_kernel<<<B, BDIM, 0, stream>>>(
                gi_f, gh_f, h_f + (size_t)i * B * H, xin_f,
                ln_s + (size_t)i * H, ln_b + (size_t)i * H,
                h_f + (size_t)i * B * H, h_bf + (size_t)i * B * H);
        }

        // y = relu(h2 @ fc1W^T + fc1b)
        gemm_bf16_wmma_kernel<<<dim3(2 * H / 128, B / 128), BDIM, 0, stream>>>(
            h_bf + (size_t)2 * B * H, H, nullptr, fc1W_bf, H, fc1b,
            nullptr, y1_bf, 2 * H, H, 1);
        // logits = y @ fc2W^T + fc2b
        gemm_bf16_wmma_kernel<<<dim3(V / 128, B / 128), BDIM, 0, stream>>>(
            y1_bf, 2 * H, nullptr, fc2W_bf, 2 * H, fc2b,
            logits, nullptr, V, 2 * H, 0);
        // acc += log_softmax(logits)[target_t]
        logsm_acc_kernel<<<B, BDIM, 0, stream>>>(logits, target, t, out);
    }
}
